// GhostLinearTandem_50663434223826
// MI455X (gfx1250) — matrix operand order repaired and checked by reading, not compiled, not run on hardware
//
#include <hip/hip_runtime.h>

// GhostLinearTandem fused dequant-GEMM for gfx1250 (MI455X).
// C[512,8192] = X[512,4096] * W[8192,4096]^T,  W[o][i] = lut[base*256+fine]*scale[o]
// - index tensors (268 MB) read exactly once: each block covers full M=512
// - dequant -> f16 in LDS, GEMM via v_wmma_f32_16x16x32_f16, f32 accumulate

#define IN_F   4096
#define OUT_F  8192
#define M_TOT  512
#define BN     64     // output features per block
#define BK     32     // K per step (one WMMA K)
#define XPAD   40     // padded LDS row length (f16 elems) for x tile
#define WPAD   40     // padded LDS row length (f16 elems) for w tile

typedef __attribute__((ext_vector_type(16))) _Float16 v16h;
typedef __attribute__((ext_vector_type(8)))  _Float16 v8h;
typedef __attribute__((ext_vector_type(4)))  _Float16 v4h;
typedef __attribute__((ext_vector_type(8)))  float    v8f;
typedef __attribute__((ext_vector_type(4)))  float    f32x4;
typedef __attribute__((ext_vector_type(4)))  int      i32x4;

__global__ __launch_bounds__(256)
void ghost_linear_tandem_kernel(const float* __restrict__ x,
                                const int*   __restrict__ base_idx,
                                const int*   __restrict__ fine_idx,
                                const float* __restrict__ scale,
                                const float* __restrict__ lut,
                                float*       __restrict__ out)
{
    __shared__ _Float16 x_tile[M_TOT * XPAD];   // 512*40*2 = 40 KB
    __shared__ _Float16 w_tile[BN   * WPAD];    //  64*40*2 =  5 KB

    const int tid  = threadIdx.x;
    const int lane = tid & 31;
    const int wave = tid >> 5;              // 0..7
    const int n0   = blockIdx.x * BN;       // output-feature base of this block
    const int m0w  = wave * 64;             // each wave: 64 rows x 64 cols

    // per-wave accumulators: 4 M-tiles x 4 N-tiles of 16x16 f32
    v8f acc[4][4];
    #pragma unroll
    for (int i = 0; i < 4; ++i)
        #pragma unroll
        for (int j = 0; j < 4; ++j)
            acc[i][j] = (v8f){0.f,0.f,0.f,0.f,0.f,0.f,0.f,0.f};

    // dequant work split: thread -> (o = tid>>2, 8 consecutive k at (tid&3)*8)
    const int dq_o  = tid >> 2;             // 0..63
    const int dq_kc = (tid & 3) * 8;        // 0,8,16,24
    const float dq_s = scale[n0 + dq_o];
    const int* bp = base_idx + (size_t)(n0 + dq_o) * IN_F + dq_kc;
    const int* fp = fine_idx + (size_t)(n0 + dq_o) * IN_F + dq_kc;

    // fragment geometry (ISA 16-bit A 16x32 / B 32x16 lane layouts)
    const int frow = lane & 15;
    const int ak   = (lane >> 4) * 8;       // A: K chunks {ak..ak+7, 16+ak..16+ak+7}
    const int bk   = (lane >> 4) * 16;      // unused; B reads the same K chunks as A, at ak and 16+ak

    for (int k0 = 0; k0 < IN_F; k0 += BK) {
        __syncthreads();   // LDS reuse fence vs previous step's compute

        // ---- stage X: 512 x 32 fp32 -> f16 LDS (float4 loads, v4h stores) ----
        #pragma unroll
        for (int i = 0; i < 16; ++i) {
            int slot = tid + 256 * i;        // 0..4095 float4 slots
            int row  = slot >> 3;            // 0..511
            int c4   = (slot & 7) * 4;       // 0..28
            f32x4 v = *(const f32x4*)(x + (size_t)row * IN_F + k0 + c4);
            v4h h = { (_Float16)v.x, (_Float16)v.y, (_Float16)v.z, (_Float16)v.w };
            *(v4h*)(&x_tile[row * XPAD + c4]) = h;
        }

        // ---- dequant W tile: 64 x 32 (two-level LUT + per-row scale) ----
        {
            const int* b0 = bp + k0;
            const int* f0 = fp + k0;
            i32x4 blo = *(const i32x4*)(b0);
            i32x4 bhi = *(const i32x4*)(b0 + 4);
            i32x4 flo = *(const i32x4*)(f0);
            i32x4 fhi = *(const i32x4*)(f0 + 4);
            // prefetch next step's index stream into cache
            __builtin_prefetch(b0 + BK, 0, 0);
            __builtin_prefetch(f0 + BK, 0, 0);
            v8h w;
            w[0] = (_Float16)(lut[(blo.x << 8) | flo.x] * dq_s);
            w[1] = (_Float16)(lut[(blo.y << 8) | flo.y] * dq_s);
            w[2] = (_Float16)(lut[(blo.z << 8) | flo.z] * dq_s);
            w[3] = (_Float16)(lut[(blo.w << 8) | flo.w] * dq_s);
            w[4] = (_Float16)(lut[(bhi.x << 8) | fhi.x] * dq_s);
            w[5] = (_Float16)(lut[(bhi.y << 8) | fhi.y] * dq_s);
            w[6] = (_Float16)(lut[(bhi.z << 8) | fhi.z] * dq_s);
            w[7] = (_Float16)(lut[(bhi.w << 8) | fhi.w] * dq_s);
            *(v8h*)(&w_tile[dq_o * WPAD + dq_kc]) = w;   // 16B-aligned
        }

        __syncthreads();

        // ---- WMMA: 4 M-tiles x 4 N-tiles per wave ----
        #pragma unroll
        for (int mt = 0; mt < 4; ++mt) {
            const _Float16* ar = &x_tile[(m0w + mt * 16 + frow) * XPAD];
            v8h alo = *(const v8h*)(ar + ak);
            v8h ahi = *(const v8h*)(ar + 16 + ak);
            v16h a = __builtin_shufflevector(alo, ahi,
                        0,1,2,3,4,5,6,7,8,9,10,11,12,13,14,15);
            #pragma unroll
            for (int nt = 0; nt < 4; ++nt) {
                const _Float16* br = &w_tile[(nt * 16 + frow) * WPAD + ak];
                v8h b0 = *(const v8h*)(br);
                v8h b1 = *(const v8h*)(br + 16);
                v16h b = __builtin_shufflevector(b0, b1,
                            0,1,2,3,4,5,6,7,8,9,10,11,12,13,14,15);
                acc[mt][nt] = __builtin_amdgcn_wmma_f32_16x16x32_f16(
                    false, a, false, b, (short)0, acc[mt][nt], false, false);
            }
        }
    }

    // ---- write C: 16x16 f32 layout -> row = base + r + (lane>>4)*8, col = lane&15
    const int crow = (lane >> 4) * 8;
    const int ccol = lane & 15;
    #pragma unroll
    for (int mt = 0; mt < 4; ++mt)
        #pragma unroll
        for (int nt = 0; nt < 4; ++nt)
            #pragma unroll
            for (int r = 0; r < 8; ++r) {
                int row = m0w + mt * 16 + crow + r;
                int col = n0  + nt * 16 + ccol;
                out[(size_t)row * OUT_F + col] = acc[mt][nt][r];
            }
}

extern "C" void kernel_launch(void* const* d_in, const int* in_sizes, int n_in,
                              void* d_out, int out_size, void* d_ws, size_t ws_size,
                              hipStream_t stream) {
    (void)in_sizes; (void)n_in; (void)d_ws; (void)ws_size; (void)out_size;
    const float* x        = (const float*)d_in[0];
    const int*   base_idx = (const int*)  d_in[1];
    const int*   fine_idx = (const int*)  d_in[2];
    const float* scale    = (const float*)d_in[3];
    const float* lut      = (const float*)d_in[4];
    float*       out      = (float*)d_out;

    dim3 grid(OUT_F / BN);   // 128 blocks
    dim3 block(256);         // 8 wave32 waves
    ghost_linear_tandem_kernel<<<grid, block, 0, stream>>>(
        x, base_idx, fine_idx, scale, lut, out);
}
